// CNV_Cifar10_Maj3_51943334478057
// MI455X (gfx1250) — 8-bit matrix operand order repaired and checked by reading, compiled, not run on hardware
//
#include <hip/hip_runtime.h>
#include <math.h>


typedef __attribute__((ext_vector_type(8))) int v8i;

__device__ __forceinline__ signed char fsgn(float v) {
    return (signed char)((v > 0.f) - (v < 0.f));
}

__global__ void conv1_kernel(const float* __restrict__ x,
                             const float* __restrict__ w,
                             const float* __restrict__ bias,
                             float* __restrict__ out) {
    int idx = blockIdx.x * blockDim.x + threadIdx.x;
    if (idx >= 16 * 64 * 32 * 32) return;
    int wp = idx & 31, h = (idx >> 5) & 31, co = (idx >> 10) & 63, b = idx >> 16;
    float acc = bias[co];
    for (int ci = 0; ci < 3; ++ci)
        for (int r = 0; r < 3; ++r) {
            int ih = h + r - 1;
            if (ih < 0 || ih >= 32) continue;
            for (int c = 0; c < 3; ++c) {
                int iw = wp + c - 1;
                if (iw < 0 || iw >= 32) continue;
                float wv = w[((co * 3 + ci) * 3 + r) * 3 + c];
                float ws = (wv > 0.f) ? 1.f : ((wv < 0.f) ? -1.f : 0.f);
                acc += x[((b * 3 + ci) * 32 + ih) * 32 + iw] * ws;
            }
        }
    out[idx] = acc;
}

__global__ void bn2d_stats(const float* __restrict__ xf, float* __restrict__ mean,
                           float* __restrict__ istd, int C, int HW, float eps) {
    int c = blockIdx.x, tid = threadIdx.x;
    int n = 16 * HW;
    float s = 0.f, s2 = 0.f;
    for (int i = tid; i < n; i += blockDim.x) {
        float v = xf[(((i / HW) * C) + c) * HW + (i % HW)];
        s += v; s2 += v * v;
    }
    __shared__ float sh[256], sh2[256];
    sh[tid] = s; sh2[tid] = s2; __syncthreads();
    for (int off = blockDim.x >> 1; off > 0; off >>= 1) {
        if (tid < off) { sh[tid] += sh[tid + off]; sh2[tid] += sh2[tid + off]; }
        __syncthreads();
    }
    if (tid == 0) {
        float m = sh[0] / n, v = sh2[0] / n - m * m;
        mean[c] = m; istd[c] = rsqrtf(v + eps);
    }
}

__global__ void bnsign2d(const float* __restrict__ xf, const float* __restrict__ mean,
                         const float* __restrict__ istd, const float* __restrict__ g,
                         const float* __restrict__ be, signed char* __restrict__ out,
                         int C, int HW, int total) {
    int idx = blockIdx.x * blockDim.x + threadIdx.x;
    if (idx >= total) return;
    int c = (idx / HW) % C;
    out[idx] = fsgn((xf[idx] - mean[c]) * istd[c] * g[c] + be[c]);
}

__global__ void pool_half(const int* __restrict__ gin, float* __restrict__ out,
                          int C, int H, int W, int total) {
    int idx = blockIdx.x * blockDim.x + threadIdx.x;
    if (idx >= total) return;
    int Wo = W >> 1, Ho = H >> 1;
    int wo = idx % Wo, ho = (idx / Wo) % Ho;
    int c = (idx / (Wo * Ho)) % C, b = idx / (Wo * Ho * C);
    const int* p = gin + ((b * C + c) * H + ho * 2) * W + wo * 2;
    int m = max(max(p[0], p[1]), max(p[W], p[W + 1]));
    out[idx] = 0.5f * (float)m;
}

__global__ void scale_half(const int* __restrict__ gin, float* __restrict__ out, int total) {
    int idx = blockIdx.x * blockDim.x + threadIdx.x;
    if (idx >= total) return;
    out[idx] = 0.5f * (float)gin[idx];
}


__global__ void packA_conv(const signed char* __restrict__ s, signed char* __restrict__ apk,
                           int Ci, int H, int W, int Kblk, int total) {
    int idx = blockIdx.x * blockDim.x + threadIdx.x;
    if (idx >= total) return;
    int t = idx & 3, j = (idx >> 2) & 7, lane = (idx >> 5) & 31;
    int kb = (idx >> 10) % Kblk, mt = idx / (Kblk << 10);
    int klocal = ((j >> 1) << 4) + ((lane >= 16) ? 8 : 0) + ((j & 1) << 2) + t;
    int k = kb * 64 + klocal;
    int sidx = mt * 16 + (lane & 15);
    int HW = H * W;
    int b = sidx / HW, h = (sidx % HW) / W, wq = sidx % W;
    signed char val = 0;
    int K9 = Ci * 9;
    if (k < K9) {
        int ci = k / 9, rem = k % 9, rr = rem / 3, cc = rem % 3;
        int ih = h + rr - 1, iw = wq + cc - 1;
        if (ih >= 0 && ih < H && iw >= 0 && iw < W)
            val = s[((b * Ci + ci) * H + ih) * W + iw];
    } else if (k < Ci * 12) {
        int k2 = k - K9, ci = k2 / 3, rr = k2 % 3, ih = h + rr - 1;
        if (ih >= 0 && ih < H && wq >= 1 && wq <= W - 2) {
            const signed char* sp = &s[((b * Ci + ci) * H + ih) * W + wq];
            val = (signed char)((int)sp[-1] * (int)sp[0] * (int)sp[1]);
        }
    }
    apk[idx] = val;
}

__global__ void packB_conv(const float* __restrict__ w, signed char* __restrict__ bpk,
                           int Co, int Ci, int Kblk, int total) {
    int idx = blockIdx.x * blockDim.x + threadIdx.x;
    if (idx >= total) return;
    int t = idx & 3, j = (idx >> 2) & 7, lane = (idx >> 5) & 31;
    int kb = (idx >> 10) % Kblk, nt = idx / (Kblk << 10);
    int klocal = ((j >> 1) << 4) + ((lane >= 16) ? 8 : 0) + ((j & 1) << 2) + t;
    int k = kb * 64 + klocal;
    int co = nt * 16 + (lane & 15);
    signed char val = 0;
    int K9 = Ci * 9;
    if (co < Co) {
        if (k < K9) {
            int ci = k / 9, rem = k % 9;
            val = fsgn(w[((co * Ci + ci) * 3 + rem / 3) * 3 + rem % 3]);
        } else if (k < Ci * 12) {
            int k2 = k - K9, ci = k2 / 3, rr = k2 % 3;
            const float* wp = &w[((co * Ci + ci) * 3 + rr) * 3];
            val = (signed char)(-((int)fsgn(wp[0]) * (int)fsgn(wp[1]) * (int)fsgn(wp[2])));
        }
    }
    bpk[idx] = val;
}

__global__ void packA_dense(const signed char* __restrict__ s, signed char* __restrict__ apk,
                            int K, int Kblk, int total) {
    int idx = blockIdx.x * blockDim.x + threadIdx.x;
    if (idx >= total) return;
    int t = idx & 3, j = (idx >> 2) & 7, lane = (idx >> 5) & 31;
    int kb = idx >> 10;
    int klocal = ((j >> 1) << 4) + ((lane >= 16) ? 8 : 0) + ((j & 1) << 2) + t;
    int k = kb * 64 + klocal;
    apk[idx] = (k < K) ? s[(lane & 15) * K + k] : (signed char)0;
}

__global__ void packB_dense(const float* __restrict__ w, signed char* __restrict__ bpk,
                            int N, int K, int Kblk, int total) {
    int idx = blockIdx.x * blockDim.x + threadIdx.x;
    if (idx >= total) return;
    int t = idx & 3, j = (idx >> 2) & 7, lane = (idx >> 5) & 31;
    int kb = (idx >> 10) % Kblk, nt = idx / (Kblk << 10);
    int klocal = ((j >> 1) << 4) + ((lane >= 16) ? 8 : 0) + ((j & 1) << 2) + t;
    int k = kb * 64 + klocal;
    int co = nt * 16 + (lane & 15);
    bpk[idx] = (co < N && k < K) ? fsgn(w[co * K + k]) : (signed char)0;
}

__global__ void gemm_iu8(const signed char* __restrict__ apk,
                         const signed char* __restrict__ bpk,
                         int* __restrict__ out,
                         int Kblk, int HW, int Co, int Nvalid) {
    int mt = blockIdx.x, nt = blockIdx.y;
    int lane = threadIdx.x;
    const v8i* A = (const v8i*)apk + (long)mt * Kblk * 32 + lane;
    const v8i* B = (const v8i*)bpk + (long)nt * Kblk * 32 + lane;
    v8i acc = {0, 0, 0, 0, 0, 0, 0, 0};
    for (int kb = 0; kb < Kblk; ++kb) {
        v8i a = A[kb * 32];
        v8i b = B[kb * 32];
        acc = __builtin_amdgcn_wmma_i32_16x16x64_iu8(true, a, true, b, acc, false, false);
    }
    int co = nt * 16 + (lane & 15);
    if (co < Nvalid) {
        int mhalf = (lane >= 16) ? 8 : 0;
#pragma unroll
        for (int j = 0; j < 8; ++j) {
            int srow = mt * 16 + j + mhalf;
            int b = srow / HW, p = srow % HW;
            out[((long)b * Co + co) * HW + p] = acc[j];
        }
    }
}

__global__ void bn1d_sign(const int* __restrict__ z, const float* __restrict__ g,
                          const float* __restrict__ be, signed char* __restrict__ out,
                          int N, float eps) {
    int f = blockIdx.x * blockDim.x + threadIdx.x;
    if (f >= N) return;
    float s = 0.f, s2 = 0.f;
    for (int b = 0; b < 16; ++b) { float v = (float)z[b * N + f]; s += v; s2 += v * v; }
    float m = s / 16.f, var = s2 / 16.f - m * m, is = rsqrtf(var + eps);
    for (int b = 0; b < 16; ++b)
        out[b * N + f] = fsgn(((float)z[b * N + f] - m) * is * g[f] + be[f]);
}

__global__ void final_kernel(const int* __restrict__ z, float* __restrict__ out, float eps) {
    __shared__ float zn[16][10];
    int tid = threadIdx.x;
    if (tid < 10) {
        float s = 0.f, s2 = 0.f;
        for (int b = 0; b < 16; ++b) { float v = (float)z[b * 10 + tid]; s += v; s2 += v * v; }
        float m = s / 16.f, var = s2 / 16.f - m * m, is = rsqrtf(var + eps);
        for (int b = 0; b < 16; ++b) zn[b][tid] = ((float)z[b * 10 + tid] - m) * is;
    }
    __syncthreads();
    if (tid < 16) {
        float mx = -1e30f;
        for (int j = 0; j < 10; ++j) mx = fmaxf(mx, zn[tid][j]);
        float se = 0.f;
        for (int j = 0; j < 10; ++j) se += expf(zn[tid][j] - mx);
        float lse = mx + logf(se);
        for (int j = 0; j < 10; ++j) out[tid * 10 + j] = zn[tid][j] - lse;
    }
}

extern "C" void kernel_launch(void* const* d_in, const int* in_sizes, int n_in,
                              void* d_out, int out_size, void* d_ws, size_t ws_size,
                              hipStream_t stream) {
    (void)in_sizes; (void)n_in; (void)out_size; (void)ws_size;
    const float eps = 1e-5f;
    const size_t MB = 1024 * 1024;

    const float* x  = (const float*)d_in[0];
    const float* w1 = (const float*)d_in[1];
    const float* b1 = (const float*)d_in[2];
    const float* wconv[5] = { (const float*)d_in[3], (const float*)d_in[4],
                              (const float*)d_in[5], (const float*)d_in[6],
                              (const float*)d_in[7] };
    const float* gA[6];  const float* beA[6];
    for (int i = 0; i < 6; ++i) { gA[i] = (const float*)d_in[8 + 2 * i]; beA[i] = (const float*)d_in[9 + 2 * i]; }
    const float* fw1 = (const float*)d_in[20];
    const float* fw2 = (const float*)d_in[21];
    const float* fw3 = (const float*)d_in[22];
    const float* fg1 = (const float*)d_in[23]; const float* fbe1 = (const float*)d_in[24];
    const float* fg2 = (const float*)d_in[25]; const float* fbe2 = (const float*)d_in[26];

    char* ws = (char*)d_ws;
    float*       f32buf = (float*)(ws + 0);
    int*         gout   = (int*)  (ws + 4 * MB);
    signed char* apk    = (signed char*)(ws + 8 * MB);
    signed char* bpk    = (signed char*)(ws + 21 * MB);
    float*       meanb  = (float*)(ws + 23 * MB);
    float*       istdb  = (float*)(ws + 23 * MB + 4096);
    signed char* s1  = (signed char*)(ws + 23 * MB + 64 * 1024);
    signed char* s2  = s1 + 1048576;
    signed char* s3  = s2 + 262144;
    signed char* s4  = s3 + 524288;
    signed char* s5  = s4 + 131072;
    signed char* s6  = s5 + 262144;
    signed char* sf1 = s6 + 65536;
    signed char* sf2 = sf1 + 8192;

    conv1_kernel<<<(1048576 + 255) / 256, 256, 0, stream>>>(x, w1, b1, f32buf);
    bn2d_stats<<<64, 256, 0, stream>>>(f32buf, meanb, istdb, 64, 1024, eps);
    bnsign2d<<<(1048576 + 255) / 256, 256, 0, stream>>>(f32buf, meanb, istdb,
                                                        gA[0], beA[0], s1, 64, 1024, 1048576);

    struct P { const float* w; const signed char* sin; signed char* sout;
               const float* g; const float* be; int Ci, H, W, Co, pool; };
    P ps[5] = {
        { wconv[0], s1, s2, gA[1], beA[1],  64, 32, 32,  64, 1 },
        { wconv[1], s2, s3, gA[2], beA[2],  64, 16, 16, 128, 0 },
        { wconv[2], s3, s4, gA[3], beA[3], 128, 16, 16, 128, 1 },
        { wconv[3], s4, s5, gA[4], beA[4], 128,  8,  8, 256, 0 },
        { wconv[4], s5, s6, gA[5], beA[5], 256,  8,  8, 256, 1 },
    };
    for (int L = 0; L < 5; ++L) {
        P& p = ps[L];
        int HW = p.H * p.W;
        int Mt = HW;
        int Nt = p.Co / 16;
        int Kb = (p.Ci * 12) / 64;
        int totB = Nt * Kb * 1024, totA = Mt * Kb * 1024;
        packB_conv<<<(totB + 255) / 256, 256, 0, stream>>>(p.w, bpk, p.Co, p.Ci, Kb, totB);
        packA_conv<<<(totA + 255) / 256, 256, 0, stream>>>(p.sin, apk, p.Ci, p.H, p.W, Kb, totA);
        gemm_iu8<<<dim3(Mt, Nt), 32, 0, stream>>>(apk, bpk, gout, Kb, HW, p.Co, p.Co);
        int Ho = p.pool ? p.H / 2 : p.H, Wo = p.pool ? p.W / 2 : p.W;
        int totO = 16 * p.Co * Ho * Wo;
        if (p.pool)
            pool_half<<<(totO + 255) / 256, 256, 0, stream>>>(gout, f32buf, p.Co, p.H, p.W, totO);
        else
            scale_half<<<(totO + 255) / 256, 256, 0, stream>>>(gout, f32buf, totO);
        bn2d_stats<<<p.Co, 256, 0, stream>>>(f32buf, meanb, istdb, p.Co, Ho * Wo, eps);
        bnsign2d<<<(totO + 255) / 256, 256, 0, stream>>>(f32buf, meanb, istdb,
                                                         p.g, p.be, p.sout, p.Co, Ho * Wo, totO);
    }

    { int Kb = 64, Nt = 32, totB = Nt * Kb * 1024, totA = Kb * 1024;
      packB_dense<<<(totB + 255) / 256, 256, 0, stream>>>(fw1, bpk, 512, 4096, Kb, totB);
      packA_dense<<<(totA + 255) / 256, 256, 0, stream>>>(s6, apk, 4096, Kb, totA);
      gemm_iu8<<<dim3(1, Nt), 32, 0, stream>>>(apk, bpk, gout, Kb, 1, 512, 512);
      bn1d_sign<<<2, 256, 0, stream>>>(gout, fg1, fbe1, sf1, 512, eps); }

    { int Kb = 8, Nt = 32, totB = Nt * Kb * 1024, totA = Kb * 1024;
      packB_dense<<<(totB + 255) / 256, 256, 0, stream>>>(fw2, bpk, 512, 512, Kb, totB);
      packA_dense<<<(totA + 255) / 256, 256, 0, stream>>>(sf1, apk, 512, Kb, totA);
      gemm_iu8<<<dim3(1, Nt), 32, 0, stream>>>(apk, bpk, gout, Kb, 1, 512, 512);
      bn1d_sign<<<2, 256, 0, stream>>>(gout, fg2, fbe2, sf2, 512, eps); }

    { int Kb = 8, totB = Kb * 1024, totA = Kb * 1024;
      packB_dense<<<(totB + 255) / 256, 256, 0, stream>>>(fw3, bpk, 10, 512, Kb, totB);
      packA_dense<<<(totA + 255) / 256, 256, 0, stream>>>(sf2, apk, 512, Kb, totA);
      gemm_iu8<<<dim3(1, 1), 32, 0, stream>>>(apk, bpk, gout, Kb, 1, 10, 10);
      final_kernel<<<1, 32, 0, stream>>>(gout, (float*)d_out, eps); }
}
